// StitchAttention_62277025792798
// MI455X (gfx1250) — compile-verified, matrix operand order checked by reading, not run on hardware
//
#include <hip/hip_runtime.h>
#include <hip/hip_bf16.h>

// ---------------------------------------------------------------------------
// StitchAttention on MI455X (gfx1250, wave32, WMMA)
//   QKV 1x1 convs and the 3x3 conv (as 9 shifted 1x1 GEMMs) run on
//   v_wmma_f32_16x16x32_bf16 with fp32 accumulate.  Fragments are fed with
//   wide loads: A = 2x ds_load_b128 from a padded M-major LDS tile
//   (conflict-free: row stride 100 dwords), hoisted once per k-step per wave
//   (all 6 tiles of a wave share the same M row); B = 2x global_load_b128
//   from K-contiguous bf16 weights (L2-resident).  Attention (seq len
//   4/9/16, head dim up to 147k) is bandwidth bound -> fp32 reductions.
// ---------------------------------------------------------------------------

typedef __attribute__((ext_vector_type(16))) __bf16 v16bf;
typedef __attribute__((ext_vector_type(8)))  __bf16 v8bf;
typedef __attribute__((ext_vector_type(8)))  float  v8f;
typedef __attribute__((ext_vector_type(4)))  float  v4f;

#define BATCH 16
#define CCH   192
#define HH    96
#define WW    96
#define HWSZ  (HH * WW)                       // 9216
#define A_STRIDE 200                          // padded LDS row (bf16), 400 B
static constexpr size_t NT = (size_t)BATCH * CCH * HWSZ;   // 28,311,552

// ---- WMMA fragment helpers (CDNA5 ISA 7.12.2 16-bit layouts) ---------------
// Lane l needs K values in two contiguous runs of 8:
//   kb = k0 + 8*(l>>4);  k in [kb,kb+8) then [kb+16,kb+24)
__device__ __forceinline__ v16bf load_a_frag_lds(const __bf16* lds_a, int lane,
                                                 int mt, int k0) {
  const int m  = mt + (lane & 15);
  const int kb = k0 + ((lane >> 4) << 3);
  const __bf16* p = lds_a + m * A_STRIDE + kb;
  v8bf lo = *(const v8bf*)p;
  v8bf hi = *(const v8bf*)(p + 16);
  return __builtin_shufflevector(lo, hi, 0, 1, 2, 3, 4, 5, 6, 7,
                                 8, 9, 10, 11, 12, 13, 14, 15);
}

// B (KxN): weights stored N-major, K contiguous: wt[n*CCH + k] -> 2x b128.
__device__ __forceinline__ v16bf load_b_frag_glb(const __bf16* __restrict__ wt,
                                                 int lane, int nt, int k0) {
  const int n  = nt + (lane & 15);
  const int kb = k0 + ((lane >> 4) << 3);
  const __bf16* p = wt + (size_t)n * CCH + kb;
  v8bf lo = *(const v8bf*)p;
  v8bf hi = *(const v8bf*)(p + 16);
  return __builtin_shufflevector(lo, hi, 0, 1, 2, 3, 4, 5, 6, 7,
                                 8, 9, 10, 11, 12, 13, 14, 15);
}

// ---- weight conversion: fp32 -> bf16 ---------------------------------------
__global__ __launch_bounds__(256) void convert_weights_kernel(
    const float* __restrict__ wq, const float* __restrict__ wk,
    const float* __restrict__ wv, const float* __restrict__ w_out,
    __bf16* __restrict__ wt_qkv,   // [3][O][C]  (straight bf16 copy of [O,C])
    __bf16* __restrict__ wt_conv)  // [9][O][C]  (tap-major)
{
  const int i = blockIdx.x * 256 + threadIdx.x;
  if (i < 3 * CCH * CCH) {
    const int proj = i / (CCH * CCH);
    const int rem  = i % (CCH * CCH);
    const float* src = (proj == 0) ? wq : (proj == 1) ? wk : wv;
    wt_qkv[i] = (__bf16)src[rem];
  }
  if (i < 9 * CCH * CCH) {
    const int tap = i / (CCH * CCH);
    const int rem = i % (CCH * CCH);             // rem = o*CCH + c
    wt_conv[i] = (__bf16)w_out[(size_t)rem * 9 + tap];
  }
}

// ---- QKV projection: GEMM  [64 pix x 192] x [192 x 192] per block ----------
__global__ __launch_bounds__(256) void qkv_gemm_kernel(
    const float* __restrict__ x, const __bf16* __restrict__ wt_all,
    const float* __restrict__ bq, const float* __restrict__ bk,
    const float* __restrict__ bv,
    float* __restrict__ qo, float* __restrict__ ko, float* __restrict__ vo)
{
  const int proj = blockIdx.y;
  const int tile = blockIdx.x;                 // 0 .. 16*144-1
  const int b    = tile / 144;
  const int p0   = (tile % 144) * 64;          // pixel base within batch plane

  const __bf16* wt  = wt_all + (size_t)proj * CCH * CCH;
  const float* bias = (proj == 0) ? bq : (proj == 1) ? bk : bv;
  float*       out  = (proj == 0) ? qo : (proj == 1) ? ko : vo;

  __shared__ __align__(16) __bf16 lds_a[64 * A_STRIDE];   // 25.6 KB, M-major
  const float* xb = x + (size_t)b * CCH * HWSZ + p0;
  for (int i = threadIdx.x; i < CCH * 64; i += 256) {
    const int c = i >> 6, m = i & 63;          // coalesced global read along m
    lds_a[m * A_STRIDE + c] = (__bf16)xb[(size_t)c * HWSZ + m];
  }
  __syncthreads();

  const int wave = threadIdx.x >> 5;
  const int lane = threadIdx.x & 31;
  // each wave owns 6 consecutive tiles of a 4x12 grid: same M row, 6 N cols
  const int mt     = ((wave * 6) / 12) * 16;
  const int ntbase = ((wave * 6) % 12) * 16;

  v8f acc[6] = {};
  for (int k0 = 0; k0 < CCH; k0 += 32) {
    const v16bf a = load_a_frag_lds(lds_a, lane, mt, k0);  // hoisted: 1x / k0
#pragma unroll
    for (int j = 0; j < 6; ++j) {
      v16bf w = load_b_frag_glb(wt, lane, ntbase + j * 16, k0);
      acc[j] = __builtin_amdgcn_wmma_f32_16x16x32_bf16(
          false, a, false, w, (short)0, acc[j], false, false);
    }
  }

#pragma unroll
  for (int j = 0; j < 6; ++j) {
    const int n = ntbase + j * 16 + (lane & 15);
    const float bval = bias[n];
    float* dst = out + (size_t)(b * CCH + n) * HWSZ + p0 + mt + ((lane >> 4) << 3);
    v4f lo, hi;
#pragma unroll
    for (int r = 0; r < 4; ++r) {
      lo[r] = acc[j][r] + bval;
      hi[r] = acc[j][r + 4] + bval;
    }
    *(v4f*)dst = lo;                            // 2x global_store_b128
    *(v4f*)(dst + 4) = hi;
  }
}

// ---- attention scores: one block per (b, g, p, q) pair, LDS tree-reduce ----
__global__ __launch_bounds__(256) void scores_kernel(
    const float* __restrict__ q, const float* __restrict__ k,
    float* __restrict__ S)                      // [16][3][16][16]
{
  const int pp = blockIdx.x;                    // 0..352 = 16 + 81 + 256 pairs
  int g, hs, rel;
  if (pp < 16)       { g = 0; hs = 2; rel = pp; }
  else if (pp < 97)  { g = 1; hs = 3; rel = pp - 16; }
  else               { g = 2; hs = 4; rel = pp - 97; }
  const int np = hs * hs;
  const int p  = rel / np, pq = rel % np;
  const int pi = p / hs,  pj = p % hs;
  const int qi = pq / hs, qj = pq % hs;
  const int oh = HH / hs, ow = WW / hs;
  const int dlen = 64 * oh * ow;
  const int cb = 64 * g;
  const int b = blockIdx.y;

  float partial = 0.f;
  for (int d = threadIdx.x; d < dlen; d += 256) {
    const int cl  = d / (oh * ow);
    const int r   = d % (oh * ow);
    const int ohi = r / ow, owj = r % ow;
    const size_t base = (size_t)(b * CCH + cb + cl) * HWSZ;
    partial += q[base + (ohi * hs + pi) * WW + (owj * hs + pj)] *
               k[base + (ohi * hs + qi) * WW + (owj * hs + qj)];
  }
  __shared__ float red[256];
  red[threadIdx.x] = partial;
  __syncthreads();
  for (int s = 128; s > 0; s >>= 1) {
    if (threadIdx.x < s) red[threadIdx.x] += red[threadIdx.x + s];
    __syncthreads();
  }
  if (threadIdx.x == 0)
    S[((b * 3 + g) * 16 + p) * 16 + pq] = red[0] * rsqrtf((float)dlen);
}

// ---- softmax over tiny rows (p <= 16) --------------------------------------
__global__ void softmax_kernel(float* __restrict__ S) {
  const int g = blockIdx.x, b = blockIdx.y;
  const int hs = 2 + g, np = hs * hs;
  const int p = threadIdx.x;
  if (p < np) {
    float* row = S + ((b * 3 + g) * 16 + p) * 16;
    float mx = -3.402823466e38f;
    for (int j = 0; j < np; ++j) mx = fmaxf(mx, row[j]);
    float sum = 0.f;
    for (int j = 0; j < np; ++j) { float e = __expf(row[j] - mx); row[j] = e; sum += e; }
    const float inv = 1.f / sum;
    for (int j = 0; j < np; ++j) row[j] *= inv;
  }
}

// ---- P*V with torch-faithful unstitch layout (H=(hs,oh), W=(ws,ow)) --------
__global__ __launch_bounds__(256) void av_kernel(
    const float* __restrict__ v, const float* __restrict__ S,
    float* __restrict__ attn)
{
  const size_t idx = (size_t)blockIdx.x * 256 + threadIdx.x;
  if (idx >= NT) return;
  const int pix = (int)(idx % HWSZ);
  const size_t t = idx / HWSZ;
  const int c = (int)(t % CCH);
  const int b = (int)(t / CCH);
  const int hout = pix / WW, wout = pix % WW;
  const int g = c / 64, hs = 2 + g, oh = HH / hs;
  const int i = hout / oh, ohi = hout % oh;
  const int j = wout / oh, owj = wout % oh;     // ow == oh (square strides)
  const int p = i * hs + j, np = hs * hs;
  const float* Pr = S + ((b * 3 + g) * 16 + p) * 16;
  const size_t base = (size_t)(b * CCH + c) * HWSZ;
  float acc = 0.f;
  for (int qq = 0; qq < np; ++qq) {
    const int qi = qq / hs, qj = qq % hs;
    acc += Pr[qq] * v[base + (ohi * hs + qi) * WW + (owj * hs + qj)];
  }
  attn[idx] = acc;
}

// ---- 3x3 conv as 9 shifted 1x1 WMMA GEMMs ----------------------------------
__global__ __launch_bounds__(256) void conv_gemm_kernel(
    const float* __restrict__ attn, const __bf16* __restrict__ wt_conv,
    const float* __restrict__ b_out, float* __restrict__ ybuf)
{
  const int tile = blockIdx.x;
  const int b  = tile / 144;
  const int p0 = (tile % 144) * 64;

  __shared__ __align__(16) __bf16 lds_a[64 * A_STRIDE];
  const int wave = threadIdx.x >> 5;
  const int lane = threadIdx.x & 31;
  const int mt     = ((wave * 6) / 12) * 16;
  const int ntbase = ((wave * 6) % 12) * 16;

  v8f acc[6] = {};
  for (int tap = 0; tap < 9; ++tap) {
    const int dy = tap / 3 - 1, dx = tap % 3 - 1;
    __syncthreads();
    for (int i = threadIdx.x; i < CCH * 64; i += 256) {
      const int c = i >> 6, m = i & 63;
      const int pix = p0 + m;
      const int h = pix / WW + dy, w = pix % WW + dx;
      float val = 0.f;
      if (h >= 0 && h < HH && w >= 0 && w < WW)
        val = attn[(size_t)(b * CCH + c) * HWSZ + h * WW + w];
      lds_a[m * A_STRIDE + c] = (__bf16)val;
    }
    __syncthreads();

    const __bf16* wt = wt_conv + (size_t)tap * CCH * CCH;
    for (int k0 = 0; k0 < CCH; k0 += 32) {
      const v16bf a = load_a_frag_lds(lds_a, lane, mt, k0);
#pragma unroll
      for (int j = 0; j < 6; ++j) {
        v16bf w = load_b_frag_glb(wt, lane, ntbase + j * 16, k0);
        acc[j] = __builtin_amdgcn_wmma_f32_16x16x32_bf16(
            false, a, false, w, (short)0, acc[j], false, false);
      }
    }
  }

#pragma unroll
  for (int j = 0; j < 6; ++j) {
    const int n = ntbase + j * 16 + (lane & 15);
    const float bval = b_out[n];
    float* dst = ybuf + (size_t)(b * CCH + n) * HWSZ + p0 + mt + ((lane >> 4) << 3);
    v4f lo, hi;
#pragma unroll
    for (int r = 0; r < 4; ++r) {
      lo[r] = acc[j][r] + bval;
      hi[r] = acc[j][r + 4] + bval;
    }
    *(v4f*)dst = lo;
    *(v4f*)(dst + 4) = hi;
  }
}

// ---- batchnorm stats: one block per channel, float4 loads ------------------
__global__ __launch_bounds__(256) void bn_stats_kernel(
    const float* __restrict__ ybuf, float* __restrict__ meanv,
    float* __restrict__ invstd)
{
  const int c = blockIdx.x;
  float s = 0.f, s2 = 0.f;
  for (int i = threadIdx.x; i < BATCH * HWSZ / 4; i += 256) {
    const int b = i / (HWSZ / 4), p4 = i % (HWSZ / 4);
    const v4f y = *(const v4f*)(ybuf + (size_t)(b * CCH + c) * HWSZ + p4 * 4);
#pragma unroll
    for (int e = 0; e < 4; ++e) { s += y[e]; s2 += y[e] * y[e]; }
  }
  __shared__ float r1[256], r2[256];
  r1[threadIdx.x] = s; r2[threadIdx.x] = s2;
  __syncthreads();
  for (int st = 128; st > 0; st >>= 1) {
    if (threadIdx.x < st) {
      r1[threadIdx.x] += r1[threadIdx.x + st];
      r2[threadIdx.x] += r2[threadIdx.x + st];
    }
    __syncthreads();
  }
  if (threadIdx.x == 0) {
    const float N = (float)(BATCH * HWSZ);
    const float mean = r1[0] / N;
    const float var  = r2[0] / N - mean * mean;
    meanv[c]  = mean;
    invstd[c] = rsqrtf(var + 1e-5f);
  }
}

// ---- BN apply + LeakyReLU(0.2), float4 -------------------------------------
__global__ __launch_bounds__(256) void bn_act_kernel(
    const float* __restrict__ ybuf, const float* __restrict__ meanv,
    const float* __restrict__ invstd, const float* __restrict__ gamma,
    const float* __restrict__ beta, float* __restrict__ out)
{
  const size_t i4 = (size_t)blockIdx.x * 256 + threadIdx.x;
  if (i4 >= NT / 4) return;
  const size_t idx = i4 * 4;
  const int c = (int)((idx / HWSZ) % CCH);     // HWSZ % 4 == 0: same c for all 4
  const float sc = invstd[c] * gamma[c];
  const float sh = beta[c] - meanv[c] * sc;
  v4f y = *(const v4f*)(ybuf + idx);
  v4f r;
#pragma unroll
  for (int e = 0; e < 4; ++e) {
    const float yy = y[e] * sc + sh;
    r[e] = (yy >= 0.f) ? yy : 0.2f * yy;
  }
  *(v4f*)(out + idx) = r;
}

// ---------------------------------------------------------------------------
extern "C" void kernel_launch(void* const* d_in, const int* in_sizes, int n_in,
                              void* d_out, int out_size, void* d_ws, size_t ws_size,
                              hipStream_t stream) {
  (void)in_sizes; (void)n_in; (void)out_size; (void)ws_size;

  const float* x     = (const float*)d_in[0];
  const float* wq    = (const float*)d_in[1];
  const float* bq    = (const float*)d_in[2];
  const float* wk    = (const float*)d_in[3];
  const float* bk    = (const float*)d_in[4];
  const float* wv    = (const float*)d_in[5];
  const float* bv    = (const float*)d_in[6];
  const float* w_out = (const float*)d_in[7];
  const float* b_out = (const float*)d_in[8];
  const float* gamma = (const float*)d_in[9];
  const float* beta  = (const float*)d_in[10];
  float* out = (float*)d_out;

  // workspace carve-up
  float* q    = (float*)d_ws;
  float* kbuf = q + NT;
  float* vbuf = kbuf + NT;
  float* attn = vbuf + NT;
  float* ybuf = attn + NT;
  float* S    = ybuf + NT;            // 16*3*16*16 = 12288 floats
  float* meanv  = S + 12288;          // 192
  float* invstd = meanv + CCH;        // 192
  __bf16* wt_qkv  = (__bf16*)(invstd + CCH);      // 3*192*192
  __bf16* wt_conv = wt_qkv + 3 * CCH * CCH;       // 9*192*192

  // 1. weights -> bf16 (K-contiguous [o][c] rows)
  convert_weights_kernel<<<(9 * CCH * CCH + 255) / 256, 256, 0, stream>>>(
      wq, wk, wv, w_out, wt_qkv, wt_conv);

  // 2. QKV projections (bf16 WMMA GEMM, fp32 accumulate)
  qkv_gemm_kernel<<<dim3(BATCH * 144, 3), 256, 0, stream>>>(
      x, wt_qkv, bq, bk, bv, q, kbuf, vbuf);

  // 3. attention scores (bandwidth-bound dot reductions) + softmax
  scores_kernel<<<dim3(353, BATCH), 256, 0, stream>>>(q, kbuf, S);
  softmax_kernel<<<dim3(3, BATCH), 32, 0, stream>>>(S);

  // 4. P*V with unstitch layout
  av_kernel<<<(unsigned)((NT + 255) / 256), 256, 0, stream>>>(vbuf, S, attn);

  // 5. 3x3 conv as 9 shifted WMMA GEMMs
  conv_gemm_kernel<<<BATCH * 144, 256, 0, stream>>>(attn, wt_conv, b_out, ybuf);

  // 6. BN stats + normalize + LeakyReLU
  bn_stats_kernel<<<CCH, 256, 0, stream>>>(ybuf, meanv, invstd);
  bn_act_kernel<<<(unsigned)((NT / 4 + 255) / 256), 256, 0, stream>>>(
      ybuf, meanv, invstd, gamma, beta, out);
}
